// ExpDock_3032246911406
// MI455X (gfx1250) — matrix operand order repaired and checked by reading, not compiled, not run on hardware
//
#include <hip/hip_runtime.h>
#include <hip/hip_bf16.h>

#define BS 8
#define NSEG 1024
#define HID 512
#define NL 4
#define NKPT 10
#define KNN 9
#define NTOT (BS*2*NSEG)
#define NEDGE (NTOT*KNN)
#define NPS (2*NSEG)
#define SIGMA_C 25.0f
#define GAMMA_C 10.0f

typedef __attribute__((ext_vector_type(16))) _Float16 v16h;
typedef __attribute__((ext_vector_type(8)))  _Float16 v8h;
typedef __attribute__((ext_vector_type(8)))  float    v8f;
typedef _Float16 h16;

__device__ __forceinline__ float siluf(float x){ return x / (1.0f + __expf(-x)); }
__device__ __forceinline__ float softplusf(float x){ return (x > 20.f) ? x : log1pf(__expf(x)); }
__device__ __forceinline__ float sl1f(float x){ float ax = fabsf(x); return ax < 1.f ? 0.5f*x*x : ax-0.5f; }

__device__ __forceinline__ v8f zero_v8f(){ v8f z;
#pragma unroll
  for (int i=0;i<8;i++) z[i]=0.f;
  return z; }

__device__ __forceinline__ v8f wmma_f16(v16h a, v16h b, v8f c){
  return __builtin_amdgcn_wmma_f32_16x16x32_f16(false, a, false, b, (short)0, c, false, false);
}

// A fragment: 16(M) x 32(K) f16, row-major, row stride lda (halves, multiple of 8).
// Two aligned 16B chunks per lane -> ds_load_b128 / global_load_b128.
__device__ __forceinline__ v16h load_a_frag(const h16* A, int lda, int lane){
  int row = lane & 15;
  int kb  = (lane >> 4) * 16;
  const h16* p = A + row*lda + kb;
  v8h lo = *(const v8h*)(p);
  v8h hi = *(const v8h*)(p + 8);
  return __builtin_shufflevector(lo, hi, 0,1,2,3,4,5,6,7,8,9,10,11,12,13,14,15);
}

// B fragment from pre-packed weights: tile-major [kt][nt][lane][16], one 32B load.
__device__ __forceinline__ v16h load_bp(const h16* Wp, int kt, int nt, int lane){
  return *(const v16h*)(Wp + ((((size_t)kt*32 + nt)*32 + lane) << 4));
}

// B fragment where storage is N-major: element (k,n) at base[n*stride + k]; contiguous per lane.
__device__ __forceinline__ v16h load_bt_frag(const h16* B, int stride, int lane){
  int n    = lane & 15;
  int koff = (lane >> 4) * 16;
  return *(const v16h*)(B + (size_t)n*stride + koff);
}

__device__ __forceinline__ float det3(const float m[3][3]){
  return m[0][0]*(m[1][1]*m[2][2]-m[1][2]*m[2][1])
       - m[0][1]*(m[1][0]*m[2][2]-m[1][2]*m[2][0])
       + m[0][2]*(m[1][0]*m[2][1]-m[1][1]*m[2][0]);
}

// ---------------------------------------------------------------- small prep
__global__ void k_ms(const float* stdv, float* acc){
  if (threadIdx.x==0) acc[0] = (stdv[0]+stdv[1]+stdv[2]) * (1.f/3.f);
}

__global__ void k_cvt(const float* __restrict__ src, h16* __restrict__ dst, int n){
  int i = blockIdx.x*blockDim.x + threadIdx.x;
  if (i<n) dst[i] = (h16)src[i];
}

// Pack a K x N (N==512) f32 weight into WMMA-B fragment order:
// packed[((kt*32+nt)*32+lane)*16 + j] = W[(kt*32 + (lane>>4)*16 + j)*N + nt*16 + (lane&15)]
__global__ void k_pack_b(const float* __restrict__ W, h16* __restrict__ Wp, int K, int N){
  int i = blockIdx.x*blockDim.x + threadIdx.x;
  if (i >= K*N) return;
  int j    = i & 15;
  int lane = (i >> 4) & 31;
  int tile = i >> 9;
  int ntiles = N >> 4;
  int nt = tile % ntiles;
  int kt = tile / ntiles;
  int k = kt*32 + ((lane>>4)*16) + j;
  int n = nt*16 + (lane & 15);
  Wp[i] = (h16)W[(size_t)k*N + n];
}

__global__ void k_prep_nodes(const float* __restrict__ Xin, const float* __restrict__ center,
                             const float* __restrict__ rot, const float* __restrict__ trans,
                             const float* __restrict__ mean, const float* __restrict__ stdv,
                             const float* __restrict__ acc,
                             float* __restrict__ X, float* __restrict__ oriX, float* __restrict__ initX){
  int i = blockIdx.x*blockDim.x + threadIdx.x;
  if (i >= NTOT) return;
  int b = i / NPS;
  int side = (i % NPS) / NSEG;
  float ms = acc[0];
  float xc[3], xt[3];
  for (int d=0; d<3; d++) xc[d] = Xin[i*3+d] - center[b*3+d];
  if (side == 0){
    for (int e=0;e<3;e++){
      float s = trans[b*3+e]*ms;
      for (int d=0;d<3;d++) s += xc[d]*rot[b*9 + d*3 + e];
      xt[e]=s;
    }
  } else { xt[0]=xc[0]; xt[1]=xc[1]; xt[2]=xc[2]; }
  for (int d=0;d<3;d++){
    float xn = (xt[d]-mean[d])/stdv[d];
    X[i*3+d] = xn;
    initX[i*3+d] = xn;
    oriX[i*3+d] = (xc[d]-mean[d])/stdv[d];
  }
}

__global__ void k_prep_kp(const float* __restrict__ kpin, const float* __restrict__ center,
                          const float* __restrict__ rot, const float* __restrict__ trans,
                          const float* __restrict__ mean, const float* __restrict__ stdv,
                          const float* __restrict__ acc, float* __restrict__ kp, float* __restrict__ tkp){
  int i = blockIdx.x*blockDim.x + threadIdx.x;
  if (i >= BS*NKPT) return;
  int b = i / NKPT;
  float ms = acc[0];
  float kc[3], kt[3];
  for (int d=0;d<3;d++) kc[d] = kpin[i*3+d] - center[b*3+d];
  for (int e=0;e<3;e++){
    float s = trans[b*3+e]*ms;
    for (int d=0;d<3;d++) s += kc[d]*rot[b*9+d*3+e];
    kt[e]=s;
  }
  for (int d=0;d<3;d++){
    kp[i*3+d]  = (kc[d]-mean[d])/stdv[d];
    tkp[i*3+d] = (kt[d]-mean[d])/stdv[d];
  }
}

// ------------------------------------------------------- fused edge message
__global__ __launch_bounds__(256)
void k_edge_msg(const h16* __restrict__ Hh, const float* __restrict__ X,
                const int* __restrict__ src, const int* __restrict__ dst,
                const h16* __restrict__ W1p, const float* __restrict__ W1last,
                const h16* __restrict__ W2p, const float* __restrict__ coordW,
                float* __restrict__ agg, float* __restrict__ csum, float* __restrict__ cnt)
{
  __shared__ h16 Af[16][1056];
  __shared__ h16 Hid[16][528];
  __shared__ float s_d2[16];
  __shared__ float s_diff[16][3];
  __shared__ int   s_dst[16];
  __shared__ float s_s[16];

  int tid = threadIdx.x;
  int lane = tid & 31, wv = tid >> 5;
  int e0 = blockIdx.x * 16;

  if (tid < 16){
    int e = e0 + tid;
    int sE = src[e], dE = dst[e];
    s_dst[tid] = dE;
    float d2 = 0.f;
    for (int d=0; d<3; d++){
      float df = X[sE*3+d] - X[dE*3+d];
      s_diff[tid][d] = df;
      d2 += df*df;
    }
    s_d2[tid] = d2;
    s_s[tid] = 0.f;
  }
  // gather H features as 16B chunks
  for (int idx = tid; idx < 16*128; idx += 256){
    int e = idx >> 7, c = idx & 127;
    int c8 = c & 63;
    int node = (c < 64) ? src[e0+e] : dst[e0+e];
    *(v8h*)(&Af[e][(c < 64 ? 0 : 512) + c8*8]) = ((const v8h*)(Hh + (size_t)node*HID))[c8];
  }
  __syncthreads();

  int nt0 = wv * 4;
  v8f accv[4];
#pragma unroll
  for (int j=0;j<4;j++) accv[j] = zero_v8f();

  for (int kt=0; kt<32; kt++){
    if (kt+1 < 32)
      __builtin_prefetch((const void*)(W1p + ((((size_t)(kt+1)*32 + nt0)*32 + lane) << 4)), 0, 1);
    v16h a = load_a_frag(&Af[0][kt*32], 1056, lane);
#pragma unroll
    for (int j=0;j<4;j++)
      accv[j] = wmma_f16(a, load_bp(W1p, kt, nt0+j, lane), accv[j]);
  }
  int hi = lane >> 4, cl = lane & 15;
#pragma unroll
  for (int j=0;j<4;j++){
    int col = (nt0+j)*16 + cl;
    float wlast = W1last[col];
#pragma unroll
    for (int r=0;r<8;r++){
      int row = r + 8*hi;
      float v = accv[j][r] + s_d2[row]*wlast;
      Hid[row][col] = (h16)siluf(v);
    }
  }
  __syncthreads();

#pragma unroll
  for (int j=0;j<4;j++) accv[j] = zero_v8f();
  for (int kt=0; kt<16; kt++){
    if (kt+1 < 16)
      __builtin_prefetch((const void*)(W2p + ((((size_t)(kt+1)*32 + nt0)*32 + lane) << 4)), 0, 1);
    v16h a = load_a_frag(&Hid[0][kt*32], 528, lane);
#pragma unroll
    for (int j=0;j<4;j++)
      accv[j] = wmma_f16(a, load_bp(W2p, kt, nt0+j, lane), accv[j]);
  }
#pragma unroll
  for (int j=0;j<4;j++){
    int col = (nt0+j)*16 + cl;
    float cw = coordW[col];
#pragma unroll
    for (int r=0;r<8;r++){
      int row = r + 8*hi;
      float m = siluf(accv[j][r]);
      atomicAdd(&agg[(size_t)s_dst[row]*HID + col], m);
      atomicAdd(&s_s[row], m*cw);
    }
  }
  __syncthreads();
  if (tid < 16){
    int dE = s_dst[tid];
    float sv = s_s[tid];
    atomicAdd(&cnt[dE], 1.f);
    for (int d=0;d<3;d++) atomicAdd(&csum[dE*3+d], s_diff[tid][d]*sv);
  }
}

__global__ void k_xupd(float* __restrict__ X, const float* __restrict__ csum, const float* __restrict__ cnt){
  int i = blockIdx.x*blockDim.x + threadIdx.x;
  if (i >= NTOT) return;
  float c = cnt[i]; if (c < 1.f) c = 1.f;
  float inv = 1.f/c;
  for (int d=0;d<3;d++) X[i*3+d] += csum[i*3+d] * inv;
}

// UH = H + silu(cat(H,agg) @ upd_W)
__global__ __launch_bounds__(256)
void k_nodeupd(const h16* __restrict__ Hh, const float* __restrict__ agg,
               const float* __restrict__ H, const h16* __restrict__ Wup,
               float* __restrict__ UH)
{
  __shared__ h16 An[16][1040];
  int tid=threadIdx.x, lane=tid&31, wv=tid>>5;
  int n0 = blockIdx.x*16;
  for (int idx=tid; idx<16*64; idx+=256){
    int r = idx>>6, c8 = idx&63;
    int node = n0+r;
    *(v8h*)(&An[r][c8*8]) = ((const v8h*)(Hh + (size_t)node*HID))[c8];
    const float* ap = agg + (size_t)node*HID + c8*8;
    v8h hv;
#pragma unroll
    for (int q=0;q<8;q++) hv[q] = (h16)ap[q];
    *(v8h*)(&An[r][512 + c8*8]) = hv;
  }
  __syncthreads();
  int nt0 = wv*4;
  v8f accv[4];
#pragma unroll
  for (int j=0;j<4;j++) accv[j]=zero_v8f();
  for (int kt=0; kt<32; kt++){
    if (kt+1 < 32)
      __builtin_prefetch((const void*)(Wup + ((((size_t)(kt+1)*32 + nt0)*32 + lane) << 4)), 0, 1);
    v16h a = load_a_frag(&An[0][kt*32], 1040, lane);
#pragma unroll
    for (int j=0;j<4;j++)
      accv[j] = wmma_f16(a, load_bp(Wup, kt, nt0+j, lane), accv[j]);
  }
  int hi=lane>>4, cl=lane&15;
#pragma unroll
  for (int j=0;j<4;j++){
    int col = (nt0+j)*16 + cl;
#pragma unroll
    for (int r=0;r<8;r++){
      int row = r + 8*hi; int node = n0+row;
      UH[(size_t)node*HID+col] = H[(size_t)node*HID+col] + siluf(accv[j][r]);
    }
  }
}

__global__ void k_means(const float* __restrict__ V, float* __restrict__ out){
  int i = blockIdx.x*blockDim.x + threadIdx.x;
  if (i >= BS*2*HID) return;
  int h = i % HID; int bs2 = i / HID;
  int b = bs2 >> 1, side = bs2 & 1;
  const float* p = V + ((size_t)(b*NPS + side*NSEG))*HID + h;
  float s=0;
  for (int n=0;n<NSEG;n++) s += p[(size_t)n*HID];
  out[i] = s * (1.f/NSEG);
}

__global__ __launch_bounds__(256)
void k_attn(const float* __restrict__ UH, const float* __restrict__ means, float* __restrict__ anorm){
  int b = blockIdx.x, side = blockIdx.y;
  __shared__ float lg[NSEG];
  __shared__ float red[256];
  const float* mv = means + (b*2 + (1-side))*HID;
  int base = b*NPS + side*NSEG;
  int tid = threadIdx.x;
  for (int n=tid; n<NSEG; n+=256){
    const float* u = UH + (size_t)(base+n)*HID;
    float s=0;
    for (int h=0;h<HID;h++) s += u[h]*mv[h];
    lg[n]=s;
  }
  __syncthreads();
  float m = -1e30f;
  for (int n=tid;n<NSEG;n+=256) m = fmaxf(m, lg[n]);
  red[tid]=m; __syncthreads();
  for (int s=128;s>0;s>>=1){ if(tid<s) red[tid]=fmaxf(red[tid],red[tid+s]); __syncthreads(); }
  m = red[0]; __syncthreads();
  float sum=0;
  for (int n=tid;n<NSEG;n+=256){ float e=__expf(lg[n]-m); lg[n]=e; sum+=e; }
  red[tid]=sum; __syncthreads();
  for (int s=128;s>0;s>>=1){ if(tid<s) red[tid]+=red[tid+s]; __syncthreads(); }
  float inv = 1.f/red[0];
  for (int n=tid;n<NSEG;n+=256) anorm[base+n] = lg[n]*inv;
}

// H = U + (silu((U*a)@ia_W1 + b1) @ ia_W2 + b2)
__global__ __launch_bounds__(256)
void k_mlp(const float* __restrict__ UH, const float* __restrict__ anorm,
           const h16* __restrict__ Wa, const float* __restrict__ b1,
           const h16* __restrict__ Wb, const float* __restrict__ b2,
           float* __restrict__ H)
{
  __shared__ h16 hin[16][528];
  __shared__ h16 tmid[16][528];
  int tid=threadIdx.x, lane=tid&31, wv=tid>>5;
  int n0 = blockIdx.x*16;
  for (int idx=tid; idx<16*64; idx+=256){
    int r=idx>>6, c8=idx&63;
    int node=n0+r;
    float an = anorm[node];
    const float* up = UH + (size_t)node*HID + c8*8;
    v8h hv;
#pragma unroll
    for (int q=0;q<8;q++) hv[q] = (h16)(up[q]*an);
    *(v8h*)(&hin[r][c8*8]) = hv;
  }
  __syncthreads();
  int nt0 = wv*4;
  v8f accv[4];
#pragma unroll
  for (int j=0;j<4;j++) accv[j]=zero_v8f();
  for (int kt=0; kt<16; kt++){
    if (kt+1 < 16)
      __builtin_prefetch((const void*)(Wa + ((((size_t)(kt+1)*32 + nt0)*32 + lane) << 4)), 0, 1);
    v16h a = load_a_frag(&hin[0][kt*32], 528, lane);
#pragma unroll
    for (int j=0;j<4;j++)
      accv[j] = wmma_f16(a, load_bp(Wa, kt, nt0+j, lane), accv[j]);
  }
  int hi=lane>>4, cl=lane&15;
#pragma unroll
  for (int j=0;j<4;j++){
    int col = (nt0+j)*16 + cl;
    float bias = b1[col];
#pragma unroll
    for (int r=0;r<8;r++){
      int row = r + 8*hi;
      tmid[row][col] = (h16)siluf(accv[j][r]+bias);
    }
  }
  __syncthreads();
#pragma unroll
  for (int j=0;j<4;j++) accv[j]=zero_v8f();
  for (int kt=0; kt<16; kt++){
    if (kt+1 < 16)
      __builtin_prefetch((const void*)(Wb + ((((size_t)(kt+1)*32 + nt0)*32 + lane) << 4)), 0, 1);
    v16h a = load_a_frag(&tmid[0][kt*32], 528, lane);
#pragma unroll
    for (int j=0;j<4;j++)
      accv[j] = wmma_f16(a, load_bp(Wb, kt, nt0+j, lane), accv[j]);
  }
#pragma unroll
  for (int j=0;j<4;j++){
    int col = (nt0+j)*16 + cl;
    float bias = b2[col];
#pragma unroll
    for (int r=0;r<8;r++){
      int row = r + 8*hi; int node=n0+row;
      float u = UH[(size_t)node*HID+col];
      H[(size_t)node*HID+col] = u + accv[j][r] + bias;
    }
  }
}

// ----------------------------------------------------- keypoint attention
__global__ void k_vkp(const float* __restrict__ w1, const float* __restrict__ w2,
                      const float* __restrict__ Hm, float* __restrict__ v1, float* __restrict__ v2){
  int bk = blockIdx.x;
  int which = blockIdx.y;
  int b = bk / NKPT, k = bk % NKPT;
  const float* W = (which==0 ? w1 : w2) + (size_t)k*HID*HID;
  const float* g = Hm + (b*2 + (which==0 ? 1 : 0))*HID;
  float* out = (which==0 ? v1 : v2) + (size_t)bk*HID;
  for (int h = threadIdx.x; h < HID; h += blockDim.x){
    const float* wr = W + (size_t)h*HID;
    float s=0;
    for (int gg=0; gg<HID; gg++) s += wr[gg]*g[gg];
    out[h]=s;
  }
}

__global__ void k_score(const float* __restrict__ H, const float* __restrict__ v1,
                        const float* __restrict__ v2, float* __restrict__ s1, float* __restrict__ s2){
  int which = blockIdx.y;
  int i = blockIdx.x*blockDim.x + threadIdx.x;
  if (i >= BS*NKPT*NSEG) return;
  int n = i % NSEG; int bk = i / NSEG;
  int b = bk / NKPT;
  int node = b*NPS + which*NSEG + n;
  const float* v = (which==0?v1:v2) + (size_t)bk*HID;
  const float* hrow = H + (size_t)node*HID;
  float s=0;
  for (int h=0;h<HID;h++) s += hrow[h]*v[h];
  float sc = rsqrtf((float)HID);
  ((which==0)?s1:s2)[i] = s*sc;
}

__global__ __launch_bounds__(256)
void k_softy(const float* __restrict__ s1, const float* __restrict__ s2,
             const float* __restrict__ X, float* __restrict__ Y1, float* __restrict__ Y2){
  int bk = blockIdx.x; int which = blockIdx.y;
  int b = bk / NKPT;
  const float* sp_ = (which==0?s1:s2) + (size_t)bk*NSEG;
  float* Y = (which==0?Y1:Y2) + (size_t)bk*3;
  __shared__ float w[NSEG];
  __shared__ float red[256];
  int tid=threadIdx.x;
  float m=-1e30f;
  for (int n=tid;n<NSEG;n+=256) m=fmaxf(m, sp_[n]);
  red[tid]=m; __syncthreads();
  for (int s=128;s>0;s>>=1){ if(tid<s) red[tid]=fmaxf(red[tid],red[tid+s]); __syncthreads(); }
  m=red[0]; __syncthreads();
  float sum=0;
  for (int n=tid;n<NSEG;n+=256){ float e=__expf(sp_[n]-m); w[n]=e; sum+=e; }
  red[tid]=sum; __syncthreads();
  for (int s=128;s>0;s>>=1){ if(tid<s) red[tid]+=red[tid+s]; __syncthreads(); }
  float inv=1.f/red[0];
  __syncthreads();
  for (int d=0; d<3; d++){
    float p=0;
    for (int n=tid;n<NSEG;n+=256){
      int node = b*NPS + which*NSEG + n;
      p += w[n]*X[node*3+d];
    }
    red[tid]=p; __syncthreads();
    for (int s=128;s>0;s>>=1){ if(tid<s) red[tid]+=red[tid+s]; __syncthreads(); }
    if (tid==0) Y[d]=red[0]*inv;
    __syncthreads();
  }
}

// ----------------------------------------------------------------- losses
__global__ void k_ot(const float* __restrict__ Y1, const float* __restrict__ Y2,
                     const float* __restrict__ tkp, const float* __restrict__ kp, float* __restrict__ acc){
  int t = threadIdx.x;
  if (t >= 2*BS*NKPT) return;
  int which = t / (BS*NKPT);
  int bk = t % (BS*NKPT);
  int b = bk / NKPT;
  const float* Y = (which==0?Y1:Y2) + bk*3;
  const float* T = (which==0?tkp:kp) + b*NKPT*3;
  float best=1e30f; int bj=0;
  for (int j=0;j<NKPT;j++){
    float d2=0;
    for (int d=0;d<3;d++){ float df=Y[d]-T[j*3+d]; d2+=df*df; }
    if (d2<best){best=d2; bj=j;}
  }
  float e=0;
  for (int d=0;d<3;d++){ float df=Y[d]-T[bj*3+d]; e+=df*df; }
  atomicAdd(&acc[1+which], e);
}

__global__ void k_kabsch(const float* __restrict__ Y1, const float* __restrict__ Y2,
                         float* __restrict__ Rm, float* __restrict__ tv){
  int b = blockIdx.x;
  if (threadIdx.x != 0) return;
  const float* A = Y1 + b*NKPT*3;
  const float* B = Y2 + b*NKPT*3;
  float c1[3]={0,0,0}, c2[3]={0,0,0};
  for (int i=0;i<NKPT;i++) for(int d=0;d<3;d++){ c1[d]+=A[i*3+d]; c2[d]+=B[i*3+d]; }
  for (int d=0;d<3;d++){ c1[d]*= (1.f/NKPT); c2[d]*=(1.f/NKPT); }
  float M[3][3]={{0,0,0},{0,0,0},{0,0,0}};
  for (int i=0;i<NKPT;i++)
    for (int r=0;r<3;r++) for (int cc=0;cc<3;cc++)
      M[r][cc] += (A[i*3+r]-c1[r])*(B[i*3+cc]-c2[cc]);
  float S[3][3];
  for (int r=0;r<3;r++) for (int cc=0;cc<3;cc++){
    float s=0; for (int k=0;k<3;k++) s += M[k][r]*M[k][cc];
    S[r][cc]=s;
  }
  float V[3][3]={{1,0,0},{0,1,0},{0,0,1}};
  const int JP[3]={0,0,1}, JQ[3]={1,2,2};
  for (int sweep=0; sweep<30; sweep++){
    for (int pr=0; pr<3; pr++){
      int p=JP[pr], q=JQ[pr];
      float apq=S[p][q];
      if (fabsf(apq) < 1e-14f) continue;
      float app=S[p][p], aqq=S[q][q];
      float tau=(aqq-app)/(2.f*apq);
      float tt = (tau>=0.f? 1.f : -1.f)/(fabsf(tau)+sqrtf(1.f+tau*tau));
      float cth = rsqrtf(1.f+tt*tt); float sth = tt*cth;
      for (int k=0;k<3;k++){
        float skp=S[k][p], skq=S[k][q];
        S[k][p]=cth*skp - sth*skq; S[k][q]=sth*skp + cth*skq;
      }
      for (int k=0;k<3;k++){
        float spk=S[p][k], sqk=S[q][k];
        S[p][k]=cth*spk - sth*sqk; S[q][k]=sth*spk + cth*sqk;
      }
      for (int k=0;k<3;k++){
        float vp=V[k][p], vq=V[k][q];
        V[k][p]=cth*vp - sth*vq; V[k][q]=sth*vp + cth*vq;
      }
    }
  }
  float ev[3]={S[0][0],S[1][1],S[2][2]};
  int ord[3]={0,1,2};
  for (int i=0;i<2;i++) for (int j=i+1;j<3;j++)
    if (ev[ord[j]]>ev[ord[i]]){int t=ord[i];ord[i]=ord[j];ord[j]=t;}
  float U[3][3], Vs[3][3], sg[3];
  for (int c=0;c<3;c++){
    int j = ord[c];
    float s = sqrtf(fmaxf(ev[j],0.f));
    sg[c]=s;
    for (int r=0;r<3;r++){
      float u=0; for (int k=0;k<3;k++) u += M[r][k]*V[k][j];
      U[r][c] = (s > 1e-8f) ? u/s : u;
      Vs[r][c] = V[r][j];
    }
  }
  if (sg[2] <= 1e-8f){
    U[0][2]=U[1][0]*U[2][1]-U[2][0]*U[1][1];
    U[1][2]=U[2][0]*U[0][1]-U[0][0]*U[2][1];
    U[2][2]=U[0][0]*U[1][1]-U[1][0]*U[0][1];
  }
  float d = (det3(U)*det3(Vs) < 0.f) ? -1.f : 1.f;
  float R_[3][3];
  for (int r=0;r<3;r++) for (int c=0;c<3;c++)
    R_[r][c] = U[r][0]*Vs[c][0] + U[r][1]*Vs[c][1] + d*U[r][2]*Vs[c][2];
  for (int r=0;r<3;r++) for (int c=0;c<3;c++) Rm[b*9+r*3+c]=R_[r][c];
  for (int j=0;j<3;j++)
    tv[b*3+j] = c2[j] - (c1[0]*R_[0][j]+c1[1]*R_[1][j]+c1[2]*R_[2][j]);
}

__global__ void k_dock(const float* __restrict__ rot, const float* __restrict__ trans,
                       const float* __restrict__ mean, const float* __restrict__ Rm,
                       const float* __restrict__ tv, float* __restrict__ acc){
  int b = threadIdx.x;
  if (b >= BS) return;
  float ms = acc[0];
  float e1=0;
  for (int i=0;i<3;i++) for (int j=0;j<3;j++){
    float s=0; for (int k=0;k<3;k++) s += rot[b*9+i*3+k]*Rm[b*9+k*3+j];
    float d = s - (i==j?1.f:0.f);
    e1 += d*d;
  }
  float e2=0;
  for (int k=0;k<3;k++){
    float s=0;
    for (int j=0;j<3;j++) s += (trans[b*3+j]-mean[j]/ms)*rot[b*9+k*3+j];
    float e = s + tv[b*3+k] + mean[k]/ms;
    e2 += e*e;
  }
  atomicAdd(&acc[3], e1);
  atomicAdd(&acc[4], e2);
}

__global__ __launch_bounds__(256)
void k_stable(const float* __restrict__ Y1, const float* __restrict__ Y2,
              const float* __restrict__ initX, float* __restrict__ acc){
  int b = blockIdx.x, side = blockIdx.y;
  const float* Y = (side==0?Y1:Y2) + b*NKPT*3;
  const float* P = initX + (size_t)(b*NPS + side*NSEG)*3;
  float ms = acc[0];
  __shared__ float red[256];
  int tid = threadIdx.x;
  float ma = -1e30f;
  if (tid < 120){
    int ia=0, ib=0, ic=0, cntc=0, found=0;
    for (int i=0;i<NKPT && !found;i++)
      for (int j=i+1;j<NKPT && !found;j++)
        for (int k=j+1;k<NKPT && !found;k++){
          if (cntc==tid){ ia=i; ib=j; ic=k; found=1; }
          cntc++;
        }
    float u[3],v[3];
    for (int d=0;d<3;d++){ u[d]=Y[ib*3+d]-Y[ia*3+d]; v[d]=Y[ic*3+d]-Y[ia*3+d]; }
    float cx=u[1]*v[2]-u[2]*v[1], cy=u[2]*v[0]-u[0]*v[2], cz=u[0]*v[1]-u[1]*v[0];
    ma = 0.5f*sqrtf(cx*cx+cy*cy+cz*cz);
  }
  red[tid]=ma; __syncthreads();
  for (int s=128;s>0;s>>=1){ if(tid<s) red[tid]=fmaxf(red[tid],red[tid+s]); __syncthreads(); }
  float maxarea = red[0];
  __syncthreads();
  float pen=0;
  for (int k=0;k<NKPT;k++){
    float q0=Y[k*3+0]*ms, q1=Y[k*3+1]*ms, q2=Y[k*3+2]*ms;
    float p_=0;
    for (int n=tid;n<NSEG;n+=256){
      float d0=q0-P[n*3+0]*ms, d1=q1-P[n*3+1]*ms, d2_=q2-P[n*3+2]*ms;
      float d2v = d0*d0+d1*d1+d2_*d2_;
      p_ += __expf(-d2v*(1.f/SIGMA_C));
    }
    red[tid]=p_; __syncthreads();
    for (int s=128;s>0;s>>=1){ if(tid<s) red[tid]+=red[tid+s]; __syncthreads(); }
    if (tid==0){
      float G = -SIGMA_C*logf(fmaxf(red[0],1e-30f));
      pen += fmaxf(GAMMA_C - G, 0.f);
    }
    __syncthreads();
  }
  if (tid==0){
    float contrib = softplusf(-maxarea) + (pen*(1.f/NKPT))/(ms*ms);
    atomicAdd(&acc[5], contrib);
  }
}

// p = H1 H2^T tile (16x16) via WMMA, fused masked smooth-L1 softplus reduction
__global__ __launch_bounds__(32)
void k_match(const h16* __restrict__ Hh, const float* __restrict__ oriX,
             const float* __restrict__ acc, float* __restrict__ accB){
  int b = blockIdx.z;
  int n0 = blockIdx.x*16, m0 = blockIdx.y*16;
  int lane = threadIdx.x;
  __shared__ float ox0[16][3], ox1[16][3];
  __shared__ float redp[32][4];
  if (lane < 16){
    int node = b*NPS + n0 + lane;
    for (int d=0;d<3;d++) ox0[lane][d] = oriX[node*3+d];
  } else {
    int node = b*NPS + NSEG + m0 + (lane-16);
    for (int d=0;d<3;d++) ox1[lane-16][d] = oriX[node*3+d];
  }
  __syncthreads();
  const h16* H1p = Hh + (size_t)(b*NPS + n0)*HID;
  const h16* H2p = Hh + (size_t)(b*NPS + NSEG + m0)*HID;
  v8f c = zero_v8f();
  for (int kt=0; kt<16; kt++){
    v16h a  = load_a_frag(H1p + kt*32, HID, lane);
    v16h bb = load_bt_frag(H2p + kt*32, HID, lane);
    c = wmma_f16(a, bb, c);
  }
  float ms = acc[0];
  float thr = (8.f/ms); thr *= thr;
  int hi = lane>>4, cl = lane&15;
  float sMir=0, cMir=0, sDk=0, cDk=0;
#pragma unroll
  for (int r=0;r<8;r++){
    int row = r + 8*hi;
    float p = c[r];
    float d2v=0;
    for (int d=0;d<3;d++){ float df=ox0[row][d]-ox1[cl][d]; d2v+=df*df; }
    if (d2v < thr){ sDk += sl1f(softplusf(-p)); cDk += 1.f; }
    else          { sMir += sl1f(softplusf(p)); cMir += 1.f; }
  }
  redp[lane][0]=sMir; redp[lane][1]=cMir; redp[lane][2]=sDk; redp[lane][3]=cDk;
  __syncthreads();
  if (lane==0){
    float t0=0,t1=0,t2=0,t3=0;
    for (int i=0;i<32;i++){ t0+=redp[i][0]; t1+=redp[i][1]; t2+=redp[i][2]; t3+=redp[i][3]; }
    atomicAdd(&accB[b*4+0], t0);
    atomicAdd(&accB[b*4+1], t1);
    atomicAdd(&accB[b*4+2], t2);
    atomicAdd(&accB[b*4+3], t3);
  }
}

__global__ void k_rmsd(const float* __restrict__ initX, const float* __restrict__ oriX,
                       const float* __restrict__ Rm, const float* __restrict__ tv, float* __restrict__ acc){
  int i = blockIdx.x*blockDim.x + threadIdx.x;
  if (i >= BS*NSEG) return;
  int b = i / NSEG, n = i % NSEG;
  int node = b*NPS + n;
  float e=0;
  for (int ee=0; ee<3; ee++){
    float s = tv[b*3+ee];
    for (int d=0;d<3;d++) s += initX[node*3+d]*Rm[b*9+d*3+ee];
    float df = s - oriX[node*3+ee];
    e += df*df;
  }
  atomicAdd(&acc[6], e);
}

__global__ void k_final(const float* __restrict__ acc, float* __restrict__ out){
  if (threadIdx.x != 0 || blockIdx.x != 0) return;
  float ot = acc[1]*(1.f/(BS*NKPT*3)) + acc[2]*(1.f/(BS*NKPT*3));
  float dock = acc[3]/(BS*9.f) + acc[4]/(BS*3.f);
  float stable = acc[5]/(float)BS;
  float rmsd = acc[6]/((float)BS*NSEG*3);
  float match=0;
  for (int b=0;b<BS;b++){
    float cm = fmaxf(acc[8+b*4+1], 1.f);
    float cd = fmaxf(acc[8+b*4+3], 1.f);
    match += acc[8+b*4+0]/cm + acc[8+b*4+2]/cd;
  }
  match *= (1.f/BS);
  float loss = ot + dock + stable + match;
  out[0]=loss; out[1]=ot; out[2]=dock; out[3]=ot*dock; out[4]=stable; out[5]=match; out[6]=rmsd;
}

// ---------------------------------------------------------------- host
extern "C" void kernel_launch(void* const* d_in, const int* in_sizes, int n_in,
                              void* d_out, int out_size, void* d_ws, size_t ws_size,
                              hipStream_t stream)
{
  (void)in_sizes; (void)n_in; (void)out_size; (void)ws_size;
  const float* X_in   = (const float*)d_in[0];
  const float* H0     = (const float*)d_in[1];
  const float* kp_in  = (const float*)d_in[2];
  const float* center = (const float*)d_in[3];
  const float* rot    = (const float*)d_in[4];
  const float* trans  = (const float*)d_in[5];
  const float* mean   = (const float*)d_in[6];
  const float* stdv   = (const float*)d_in[7];
  const float* msgW1  = (const float*)d_in[8];
  const float* msgW2  = (const float*)d_in[9];
  const float* coordW = (const float*)d_in[10];
  const float* updW   = (const float*)d_in[11];
  const float* iaW1   = (const float*)d_in[12];
  const float* iab1   = (const float*)d_in[13];
  const float* iaW2   = (const float*)d_in[14];
  const float* iab2   = (const float*)d_in[15];
  const float* w1     = (const float*)d_in[16];
  const float* w2     = (const float*)d_in[17];
  const int*   srcI   = (const int*)d_in[18];
  const int*   dstI   = (const int*)d_in[19];
  float* out = (float*)d_out;

  float* F = (float*)d_ws;
  size_t off = 0;
  auto alloc = [&](size_t n){ size_t o = off; off += (n + 63) & ~size_t(63); return o; };
  size_t oACC = alloc(256);
  size_t oX   = alloc((size_t)NTOT*3);
  size_t oORX = alloc((size_t)NTOT*3);
  size_t oINX = alloc((size_t)NTOT*3);
  size_t oKP  = alloc(BS*NKPT*3);
  size_t oTKP = alloc(BS*NKPT*3);
  size_t oH   = alloc((size_t)NTOT*HID);
  size_t oAGG = alloc((size_t)NTOT*HID);
  size_t oUH  = alloc((size_t)NTOT*HID);
  size_t oCS  = alloc((size_t)NTOT*3);
  size_t oCN  = alloc((size_t)NTOT);
  size_t oMN  = alloc(BS*2*HID);
  size_t oAN  = alloc((size_t)NTOT);
  size_t oV1  = alloc(BS*NKPT*HID);
  size_t oV2  = alloc(BS*NKPT*HID);
  size_t oS1  = alloc((size_t)BS*NKPT*NSEG);
  size_t oS2  = alloc((size_t)BS*NKPT*NSEG);
  size_t oY1  = alloc(BS*NKPT*3);
  size_t oY2  = alloc(BS*NKPT*3);
  size_t oR   = alloc(BS*9);
  size_t oT   = alloc(BS*3);
  size_t oHH   = alloc((size_t)NTOT*HID/2);        // f16 regions (counted in floats)
  size_t oW1P  = alloc((size_t)NL*1024*HID/2);
  size_t oW2P  = alloc((size_t)NL*HID*HID/2);
  size_t oWUP  = alloc((size_t)NL*1024*HID/2);
  size_t oIA1P = alloc((size_t)NL*HID*HID/2);
  size_t oIA2P = alloc((size_t)NL*HID*HID/2);

  float* acc = F + oACC;
  h16* Hh   = (h16*)(F + oHH);
  h16* W1P  = (h16*)(F + oW1P);
  h16* W2P  = (h16*)(F + oW2P);
  h16* WUP  = (h16*)(F + oWUP);
  h16* IA1P = (h16*)(F + oIA1P);
  h16* IA2P = (h16*)(F + oIA2P);

  hipMemsetAsync(acc, 0, 256*sizeof(float), stream);
  k_ms<<<1,32,0,stream>>>(stdv, acc);
  hipMemcpyAsync(F+oH, H0, (size_t)NTOT*HID*sizeof(float), hipMemcpyDeviceToDevice, stream);
  k_prep_nodes<<<(NTOT+255)/256,256,0,stream>>>(X_in, center, rot, trans, mean, stdv, acc,
                                                F+oX, F+oORX, F+oINX);
  k_prep_kp<<<1,128,0,stream>>>(kp_in, center, rot, trans, mean, stdv, acc, F+oKP, F+oTKP);

  // pack WMMA-B weights (one tiled layout per layer)
  for (int l=0;l<NL;l++){
    k_pack_b<<<(1024*HID+255)/256,256,0,stream>>>(msgW1 + (size_t)l*1025*HID, W1P + (size_t)l*1024*HID, 1024, HID);
    k_pack_b<<<(HID*HID+255)/256,256,0,stream>>>(msgW2 + (size_t)l*HID*HID,  W2P + (size_t)l*HID*HID,  HID, HID);
    k_pack_b<<<(1024*HID+255)/256,256,0,stream>>>(updW  + (size_t)l*1024*HID, WUP + (size_t)l*1024*HID, 1024, HID);
    k_pack_b<<<(HID*HID+255)/256,256,0,stream>>>(iaW1  + (size_t)l*HID*HID,  IA1P + (size_t)l*HID*HID, HID, HID);
    k_pack_b<<<(HID*HID+255)/256,256,0,stream>>>(iaW2  + (size_t)l*HID*HID,  IA2P + (size_t)l*HID*HID, HID, HID);
  }

  for (int l=0;l<NL;l++){
    hipMemsetAsync(F+oAGG, 0, (size_t)NTOT*HID*sizeof(float), stream);
    hipMemsetAsync(F+oCS,  0, (size_t)NTOT*3*sizeof(float), stream);
    hipMemsetAsync(F+oCN,  0, (size_t)NTOT*sizeof(float), stream);
    k_cvt<<<((NTOT*HID)+255)/256,256,0,stream>>>(F+oH, Hh, NTOT*HID);
    k_edge_msg<<<NEDGE/16,256,0,stream>>>(Hh, F+oX, srcI, dstI,
        W1P + (size_t)l*1024*HID,
        msgW1 + (size_t)l*1025*HID + (size_t)1024*HID,
        W2P + (size_t)l*HID*HID,
        coordW + (size_t)l*HID,
        F+oAGG, F+oCS, F+oCN);
    k_xupd<<<NTOT/256,256,0,stream>>>(F+oX, F+oCS, F+oCN);
    k_nodeupd<<<NTOT/16,256,0,stream>>>(Hh, F+oAGG, F+oH, WUP + (size_t)l*1024*HID, F+oUH);
    k_means<<<(BS*2*HID)/256,256,0,stream>>>(F+oUH, F+oMN);
    k_attn<<<dim3(BS,2),256,0,stream>>>(F+oUH, F+oMN, F+oAN);
    k_mlp<<<NTOT/16,256,0,stream>>>(F+oUH, F+oAN,
        IA1P + (size_t)l*HID*HID, iab1 + (size_t)l*HID,
        IA2P + (size_t)l*HID*HID, iab2 + (size_t)l*HID, F+oH);
  }

  k_cvt<<<((NTOT*HID)+255)/256,256,0,stream>>>(F+oH, Hh, NTOT*HID);
  k_means<<<(BS*2*HID)/256,256,0,stream>>>(F+oH, F+oMN);
  k_vkp<<<dim3(BS*NKPT,2),256,0,stream>>>(w1, w2, F+oMN, F+oV1, F+oV2);
  k_score<<<dim3((BS*NKPT*NSEG+255)/256,2),256,0,stream>>>(F+oH, F+oV1, F+oV2, F+oS1, F+oS2);
  k_softy<<<dim3(BS*NKPT,2),256,0,stream>>>(F+oS1, F+oS2, F+oX, F+oY1, F+oY2);
  k_ot<<<1,256,0,stream>>>(F+oY1, F+oY2, F+oTKP, F+oKP, acc);
  k_kabsch<<<BS,32,0,stream>>>(F+oY1, F+oY2, F+oR, F+oT);
  k_dock<<<1,32,0,stream>>>(rot, trans, mean, F+oR, F+oT, acc);
  k_stable<<<dim3(BS,2),256,0,stream>>>(F+oY1, F+oY2, F+oINX, acc);
  k_match<<<dim3(NSEG/16, NSEG/16, BS),32,0,stream>>>(Hh, F+oORX, acc, acc+8);
  k_rmsd<<<(BS*NSEG+255)/256,256,0,stream>>>(F+oINX, F+oORX, F+oR, F+oT, acc);
  k_final<<<1,32,0,stream>>>(acc, out);
}
